// MuMURTextTransformer_8315056685641
// MI455X (gfx1250) — matrix operand order repaired and checked by reading, not compiled, not run on hardware
//
#include <hip/hip_runtime.h>
#include <hip/hip_bf16.h>

// ---------------- problem constants ----------------
#define BB   256
#define SS   77
#define DD   512
#define HH   8
#define HD   64
#define DFF  2048
#define LL   12
#define MM   (BB * SS)          // 19712 = 154 * 128
#define NEGF (-3.0e38f)

typedef __attribute__((ext_vector_type(16))) _Float16 v16h;
typedef __attribute__((ext_vector_type(8)))  float    v8f;

union HFrag { v16h v; uint4 q[2]; };

// ---------------- CDNA5 async global->LDS path (guarded) ----------------
#if __has_builtin(__builtin_amdgcn_global_load_async_to_lds_b128)
#define HAVE_ASYNC_LDS 1
typedef int v4i_ __attribute__((vector_size(16)));
typedef __attribute__((address_space(1))) v4i_* as1_v4i;
typedef __attribute__((address_space(3))) v4i_* as3_v4i;
#else
#define HAVE_ASYNC_LDS 0
#endif

// copy 16 bytes (8 halves) global -> LDS
__device__ static inline void copy_b128(const _Float16* g, _Float16* l) {
#if HAVE_ASYNC_LDS
    __builtin_amdgcn_global_load_async_to_lds_b128((as1_v4i)g, (as3_v4i)l, 0, 0);
#else
    *(uint4*)l = *(const uint4*)g;
#endif
}

template <int N>
__device__ static inline void wait_async() {
#if HAVE_ASYNC_LDS
#if __has_builtin(__builtin_amdgcn_s_wait_asynccnt)
    __builtin_amdgcn_s_wait_asynccnt((short)N);
#else
    if (N == 0) asm volatile("s_wait_asynccnt 0x0" ::: "memory");
    else        asm volatile("s_wait_asynccnt 0x4" ::: "memory");
#endif
#endif
}

// A fragment (16x32 f16, MxK): lane L holds row M=L&15; K-half selected by L>=16.
__device__ static inline v16h load_frag_a(const _Float16* base, int stride, int lane) {
    int row = lane & 15;
    int ko  = (lane >> 4) << 3;
    const _Float16* p = base + row * stride + ko;
    HFrag f;
    f.q[0] = *(const uint4*)(p);
    f.q[1] = *(const uint4*)(p + 16);
    return f.v;
}

// B fragment (32x16 f16, KxN): lane L holds col N=L&15; K-half selected by L>=16.
__device__ static inline v16h load_frag_b(const _Float16* base, int stride, int lane) {
    int n  = lane & 15;
    int ko = (lane >> 4) << 3;
    const _Float16* p = base + n * stride + ko;
    HFrag f;
    f.q[0] = *(const uint4*)(p);
    f.q[1] = *(const uint4*)(p + 16);
    return f.v;
}

__device__ static inline v8f wmma_f16(v16h a, v16h b, v8f c) {
    return __builtin_amdgcn_wmma_f32_16x16x32_f16(
        false, a, false, b, (short)0, c, false, false);
}

// ---------------- f32 -> f16 convert ----------------
__global__ __launch_bounds__(256) void cvt_kernel(const float* __restrict__ in,
                                                  _Float16* __restrict__ out, int n) {
    int i = blockIdx.x * 256 + threadIdx.x;
    if (i < n) out[i] = (_Float16)in[i];
}

// ---------------- embedding ----------------
__global__ __launch_bounds__(256) void embed_kernel(const int* __restrict__ ids,
                                                    const float* __restrict__ tok,
                                                    const float* __restrict__ pos,
                                                    float* __restrict__ x) {
    int idx = blockIdx.x * 256 + threadIdx.x;
    if (idx >= MM * DD) return;
    int d = idx & (DD - 1);
    int t = idx >> 9;
    int s = t % SS;
    int id = ids[t];
    x[idx] = tok[(size_t)id * DD + d] + pos[s * DD + d];
}

// ---------------- layernorm ----------------
__global__ __launch_bounds__(256) void ln_kernel(const float* __restrict__ x,
                                                 const float* __restrict__ w,
                                                 const float* __restrict__ b,
                                                 _Float16* __restrict__ outh,
                                                 float* __restrict__ outf) {
    __shared__ float red[256];
    int row = blockIdx.x, tid = threadIdx.x;
    const float* xr = x + (size_t)row * DD;
    float v0 = xr[tid], v1 = xr[tid + 256];
    red[tid] = v0 + v1;
    __syncthreads();
    #pragma unroll
    for (int s = 128; s > 0; s >>= 1) {
        if (tid < s) red[tid] += red[tid + s];
        __syncthreads();
    }
    float mu = red[0] * (1.0f / DD);
    __syncthreads();
    float d0 = v0 - mu, d1 = v1 - mu;
    red[tid] = d0 * d0 + d1 * d1;
    __syncthreads();
    #pragma unroll
    for (int s = 128; s > 0; s >>= 1) {
        if (tid < s) red[tid] += red[tid + s];
        __syncthreads();
    }
    float rs = rsqrtf(red[0] * (1.0f / DD) + 1e-5f);
    float y0 = d0 * rs * w[tid] + b[tid];
    float y1 = d1 * rs * w[tid + 256] + b[tid + 256];
    size_t o = (size_t)row * DD + tid;
    if (outh) { outh[o] = (_Float16)y0; outh[o + 256] = (_Float16)y1; }
    if (outf) { outf[o] = y0;           outf[o + 256] = y1; }
}

// ---------------- WMMA GEMM (double-buffered, async-staged) ----------------
// C = act((A @ W^T + bias) * scale) (+resid). A:[M,K] f16, W:[N,K] f16.
// 128x128 block tile, 8 waves, each wave 32x64 (2x4 WMMA tiles). K step 32.
#define LDS_STRIDE 40   // 32 + 8 pad halves; 80B row pitch, 16B aligned
template <int ACT, bool HASRES, bool OUTF32, bool OUTF16>
__global__ __launch_bounds__(256) void gemm_kernel(const _Float16* __restrict__ A,
                                                   const _Float16* __restrict__ W,
                                                   const float* __restrict__ bias,
                                                   const float* __restrict__ resid,
                                                   float* __restrict__ outf,
                                                   _Float16* __restrict__ outh,
                                                   int M, int N, int K,
                                                   float scale) {
    __shared__ __align__(16) _Float16 As[2][128 * LDS_STRIDE];
    __shared__ __align__(16) _Float16 Ws[2][128 * LDS_STRIDE];

    int tid  = threadIdx.x;
    int wave = tid >> 5;
    int lane = tid & 31;
    int m0 = blockIdx.y * 128;
    int n0 = blockIdx.x * 128;
    int waveM = wave >> 1;
    int waveN = wave & 1;

    v8f acc[2][4];
    #pragma unroll
    for (int mt = 0; mt < 2; ++mt)
        #pragma unroll
        for (int nt = 0; nt < 4; ++nt)
            acc[mt][nt] = (v8f){0.f, 0.f, 0.f, 0.f, 0.f, 0.f, 0.f, 0.f};

    int lr = tid >> 1;           // 0..127: staged tile row
    int lc = (tid & 1) << 4;     // 0 or 16 halves

    const _Float16* gA = A + (size_t)(m0 + lr) * K + lc;
    const _Float16* gW = W + (size_t)(n0 + lr) * K + lc;
    _Float16* lA0 = &As[0][lr * LDS_STRIDE + lc];
    _Float16* lA1 = &As[1][lr * LDS_STRIDE + lc];
    _Float16* lW0 = &Ws[0][lr * LDS_STRIDE + lc];
    _Float16* lW1 = &Ws[1][lr * LDS_STRIDE + lc];

    const int nTiles = K >> 5;

    // prologue: stage tile 0 into buffer 0
    copy_b128(gA,     lA0); copy_b128(gA + 8, lA0 + 8);
    copy_b128(gW,     lW0); copy_b128(gW + 8, lW0 + 8);

    for (int t = 0; t < nTiles; ++t) {
        int cur = t & 1;
        bool more = (t + 1) < nTiles;
        if (more) {                          // stage next tile into other buffer
            int kn = (t + 1) << 5;
            _Float16* nA = cur ? lA0 : lA1;
            _Float16* nW = cur ? lW0 : lW1;
            copy_b128(gA + kn,     nA); copy_b128(gA + kn + 8, nA + 8);
            copy_b128(gW + kn,     nW); copy_b128(gW + kn + 8, nW + 8);
            wait_async<4>();                 // current tile's 4 async ops landed
        } else {
            wait_async<0>();
        }
        __syncthreads();                     // all waves see current tile

        const _Float16* cA = As[cur];
        const _Float16* cW = Ws[cur];
        v16h af[2], bf[4];
        #pragma unroll
        for (int mt = 0; mt < 2; ++mt)
            af[mt] = load_frag_a(&cA[(waveM * 32 + mt * 16) * LDS_STRIDE], LDS_STRIDE, lane);
        #pragma unroll
        for (int nt = 0; nt < 4; ++nt)
            bf[nt] = load_frag_b(&cW[(waveN * 64 + nt * 16) * LDS_STRIDE], LDS_STRIDE, lane);
        #pragma unroll
        for (int mt = 0; mt < 2; ++mt)
            #pragma unroll
            for (int nt = 0; nt < 4; ++nt)
                acc[mt][nt] = wmma_f16(af[mt], bf[nt], acc[mt][nt]);

        __syncthreads();                     // safe to overwrite 'cur' at t+2
    }

    // epilogue: D layout -> M = j + 8*(lane>=16), N = lane&15
    int colLane = lane & 15;
    int rowHalf = (lane >> 4) << 3;
    #pragma unroll
    for (int mt = 0; mt < 2; ++mt) {
        #pragma unroll
        for (int nt = 0; nt < 4; ++nt) {
            int col  = n0 + waveN * 64 + nt * 16 + colLane;
            int rowb = m0 + waveM * 32 + mt * 16 + rowHalf;
            float bcol = bias[col];
            #pragma unroll
            for (int j = 0; j < 8; ++j) {
                int row = rowb + j;
                float y = (acc[mt][nt][j] + bcol) * scale;
                if constexpr (ACT == 1)
                    y = y / (1.0f + __expf(-1.702f * y));      // quick_gelu
                size_t off = (size_t)row * N + col;
                if constexpr (HASRES) y += resid[off];
                if constexpr (OUTF32) outf[off] = y;
                if constexpr (OUTF16) outh[off] = (_Float16)y;
            }
        }
    }
}

// ---------------- attention: one block per (b, h) ----------------
#define QK_STRIDE 72    // 64 + 8 pad halves
#define PV_STRIDE 104   // 96 + 8 pad halves
__global__ __launch_bounds__(256) void attn_kernel(const _Float16* __restrict__ q16,
                                                   const _Float16* __restrict__ k16,
                                                   const _Float16* __restrict__ v16,
                                                   const int* __restrict__ am,
                                                   _Float16* __restrict__ a16) {
    __shared__ __align__(16) _Float16 Qs[80 * QK_STRIDE];
    __shared__ __align__(16) _Float16 Ks[80 * QK_STRIDE];
    __shared__ __align__(16) _Float16 Vt[64 * PV_STRIDE];   // transposed: [d][s]
    __shared__ __align__(16) _Float16 P [80 * PV_STRIDE];   // scores -> probs

    int bh = blockIdx.x;
    int b  = bh >> 3;
    int h  = bh & 7;
    int tid = threadIdx.x, wave = tid >> 5, lane = tid & 31;
    const size_t base = (size_t)b * SS * DD + (size_t)h * HD;

    for (int i = tid; i < 80 * HD; i += 256) {
        int s = i >> 6, d = i & 63;
        _Float16 qv = (_Float16)0.f, kv = (_Float16)0.f;
        if (s < SS) {
            size_t g = base + (size_t)s * DD + d;
            qv = q16[g];
            kv = k16[g];
        }
        Qs[s * QK_STRIDE + d] = qv;
        Ks[s * QK_STRIDE + d] = kv;
    }
    for (int i = tid; i < 96 * HD; i += 256) {
        int s = i >> 6, d = i & 63;
        _Float16 vv = (_Float16)0.f;
        if (s < SS) vv = v16[base + (size_t)s * DD + d];
        Vt[d * PV_STRIDE + s] = vv;
    }
    __syncthreads();

    // scores = Q @ K^T (q pre-scaled). 5x5 tiles, K=64.
    for (int t = wave; t < 25; t += 8) {
        int qt = t / 5, kt = t % 5;
        v8f acc = (v8f){0.f, 0.f, 0.f, 0.f, 0.f, 0.f, 0.f, 0.f};
        #pragma unroll
        for (int kk = 0; kk < 64; kk += 32) {
            v16h a  = load_frag_a(&Qs[qt * 16 * QK_STRIDE + kk], QK_STRIDE, lane);
            v16h bf = load_frag_b(&Ks[kt * 16 * QK_STRIDE + kk], QK_STRIDE, lane);
            acc = wmma_f16(a, bf, acc);
        }
        int col  = kt * 16 + (lane & 15);
        int rowb = qt * 16 + ((lane >> 4) << 3);
        #pragma unroll
        for (int j = 0; j < 8; ++j)
            P[(rowb + j) * PV_STRIDE + col] = (_Float16)acc[j];
    }
    __syncthreads();

    // masked softmax (causal + padding), rows 0..76
    if (tid < 80) {
        int r = tid;
        for (int c = 80; c < 96; ++c) P[r * PV_STRIDE + c] = (_Float16)0.f;
        if (r < SS) {
            float mx = -1e30f;
            for (int c = 0; c <= r; ++c) {
                float v = (float)P[r * PV_STRIDE + c] +
                          (1.0f - (float)am[b * SS + c]) * NEGF;
                mx = fmaxf(mx, v);
            }
            float sum = 0.0f;
            for (int c = 0; c <= r; ++c) {
                float v = (float)P[r * PV_STRIDE + c] +
                          (1.0f - (float)am[b * SS + c]) * NEGF;
                float e = __expf(v - mx);
                sum += e;
                P[r * PV_STRIDE + c] = (_Float16)e;
            }
            float inv = 1.0f / sum;
            for (int c = 0; c <= r; ++c)
                P[r * PV_STRIDE + c] = (_Float16)((float)P[r * PV_STRIDE + c] * inv);
            for (int c = r + 1; c < 80; ++c) P[r * PV_STRIDE + c] = (_Float16)0.f;
        } else {
            for (int c = 0; c < 80; ++c) P[r * PV_STRIDE + c] = (_Float16)0.f;
        }
    }
    __syncthreads();

    // out = P @ V : 5x4 tiles, K=96 (zero-padded)
    for (int t = wave; t < 20; t += 8) {
        int qt = t >> 2, dt = t & 3;
        v8f acc = (v8f){0.f, 0.f, 0.f, 0.f, 0.f, 0.f, 0.f, 0.f};
        #pragma unroll
        for (int kk = 0; kk < 96; kk += 32) {
            v16h a  = load_frag_a(&P[qt * 16 * PV_STRIDE + kk], PV_STRIDE, lane);
            v16h bf = load_frag_b(&Vt[dt * 16 * PV_STRIDE + kk], PV_STRIDE, lane);
            acc = wmma_f16(a, bf, acc);
        }
        int col  = dt * 16 + (lane & 15);
        int rowb = qt * 16 + ((lane >> 4) << 3);
        #pragma unroll
        for (int j = 0; j < 8; ++j) {
            int s = rowb + j;
            if (s < SS) a16[base + (size_t)s * DD + col] = (_Float16)acc[j];
        }
    }
}

// ---------------- eot pooling ----------------
__global__ __launch_bounds__(256) void pool_kernel(const int* __restrict__ ids,
                                                   const float* __restrict__ lh,
                                                   float* __restrict__ pooled) {
    __shared__ int best;
    int b = blockIdx.x, tid = threadIdx.x;
    if (tid == 0) {
        int bi = 0, bv = ids[b * SS];
        for (int s = 1; s < SS; ++s) {
            int v = ids[b * SS + s];
            if (v > bv) { bv = v; bi = s; }
        }
        best = bi;
    }
    __syncthreads();
    const float* src = lh + ((size_t)b * SS + best) * DD;
    pooled[(size_t)b * DD + tid]       = src[tid];
    pooled[(size_t)b * DD + tid + 256] = src[tid + 256];
}

// ---------------- host orchestration ----------------
extern "C" void kernel_launch(void* const* d_in, const int* in_sizes, int n_in,
                              void* d_out, int out_size, void* d_ws, size_t ws_size,
                              hipStream_t stream) {
    const int*   ids  = (const int*)d_in[0];
    const int*   am   = (const int*)d_in[1];
    const float* tok  = (const float*)d_in[2];
    const float* pos  = (const float*)d_in[3];
    const float* ln1w = (const float*)d_in[4];
    const float* ln1b = (const float*)d_in[5];
    const float* qw   = (const float*)d_in[6];
    const float* qb   = (const float*)d_in[7];
    const float* kw   = (const float*)d_in[8];
    const float* kb   = (const float*)d_in[9];
    const float* vw   = (const float*)d_in[10];
    const float* vb   = (const float*)d_in[11];
    const float* ow   = (const float*)d_in[12];
    const float* ob   = (const float*)d_in[13];
    const float* ln2w = (const float*)d_in[14];
    const float* ln2b = (const float*)d_in[15];
    const float* f1w  = (const float*)d_in[16];
    const float* f1b  = (const float*)d_in[17];
    const float* f2w  = (const float*)d_in[18];
    const float* f2b  = (const float*)d_in[19];
    const float* lnfw = (const float*)d_in[20];
    const float* lnfb = (const float*)d_in[21];

    uintptr_t p = (uintptr_t)d_ws;
    auto carve = [&](size_t bytes) -> void* {
        void* r = (void*)p;
        p += (bytes + 255) & ~(size_t)255;
        return r;
    };
    float*    x    = (float*)   carve((size_t)MM * DD * 4);
    _Float16* h16  = (_Float16*)carve((size_t)MM * DD * 2);
    _Float16* q16  = (_Float16*)carve((size_t)MM * DD * 2);
    _Float16* k16  = (_Float16*)carve((size_t)MM * DD * 2);
    _Float16* v16  = (_Float16*)carve((size_t)MM * DD * 2);
    _Float16* a16  = (_Float16*)carve((size_t)MM * DD * 2);
    _Float16* g16  = (_Float16*)carve((size_t)MM * DFF * 2);
    _Float16* wq16 = (_Float16*)carve((size_t)DD * DD * 2);
    _Float16* wk16 = (_Float16*)carve((size_t)DD * DD * 2);
    _Float16* wv16 = (_Float16*)carve((size_t)DD * DD * 2);
    _Float16* wo16 = (_Float16*)carve((size_t)DD * DD * 2);
    _Float16* wf1  = (_Float16*)carve((size_t)DFF * DD * 2);
    _Float16* wf2  = (_Float16*)carve((size_t)DD * DFF * 2);

    float* out_lh = (float*)d_out;
    float* out_pl = out_lh + (size_t)MM * DD;

    const int nDD = DD * DD;
    const int nDF = DFF * DD;
    const dim3 gProj(DD / 128, MM / 128);
    const dim3 gFc1(DFF / 128, MM / 128);

    embed_kernel<<<(MM * DD) / 256, 256, 0, stream>>>(ids, tok, pos, x);

    const float scaleQ = 0.125f;    // HD^-0.5

    for (int l = 0; l < LL; ++l) {
        // ---- attention block ----
        ln_kernel<<<MM, 256, 0, stream>>>(x, ln1w + l * DD, ln1b + l * DD, h16, nullptr);

        cvt_kernel<<<nDD / 256, 256, 0, stream>>>(qw + (size_t)l * nDD, wq16, nDD);
        cvt_kernel<<<nDD / 256, 256, 0, stream>>>(kw + (size_t)l * nDD, wk16, nDD);
        cvt_kernel<<<nDD / 256, 256, 0, stream>>>(vw + (size_t)l * nDD, wv16, nDD);
        cvt_kernel<<<nDD / 256, 256, 0, stream>>>(ow + (size_t)l * nDD, wo16, nDD);

        gemm_kernel<0, false, false, true><<<gProj, 256, 0, stream>>>(
            h16, wq16, qb + l * DD, nullptr, nullptr, q16, MM, DD, DD, scaleQ);
        gemm_kernel<0, false, false, true><<<gProj, 256, 0, stream>>>(
            h16, wk16, kb + l * DD, nullptr, nullptr, k16, MM, DD, DD, 1.0f);
        gemm_kernel<0, false, false, true><<<gProj, 256, 0, stream>>>(
            h16, wv16, vb + l * DD, nullptr, nullptr, v16, MM, DD, DD, 1.0f);

        attn_kernel<<<BB * HH, 256, 0, stream>>>(q16, k16, v16, am, a16);

        gemm_kernel<0, true, true, false><<<gProj, 256, 0, stream>>>(
            a16, wo16, ob + l * DD, x, x, nullptr, MM, DD, DD, 1.0f);

        // ---- MLP block ----
        ln_kernel<<<MM, 256, 0, stream>>>(x, ln2w + l * DD, ln2b + l * DD, h16, nullptr);

        cvt_kernel<<<nDF / 256, 256, 0, stream>>>(f1w + (size_t)l * nDF, wf1, nDF);
        cvt_kernel<<<nDF / 256, 256, 0, stream>>>(f2w + (size_t)l * nDF, wf2, nDF);

        gemm_kernel<1, false, false, true><<<gFc1, 256, 0, stream>>>(
            h16, wf1, f1b + l * DFF, nullptr, nullptr, g16, MM, DFF, DD, 1.0f);
        gemm_kernel<0, true, true, false><<<gProj, 256, 0, stream>>>(
            g16, wf2, f2b + l * DD, x, x, nullptr, MM, DD, DFF, 1.0f);
    }

    ln_kernel<<<MM, 256, 0, stream>>>(x, lnfw, lnfb, nullptr, out_lh);
    pool_kernel<<<BB, 256, 0, stream>>>(ids, out_lh, out_pl);
}
